// RNNModel_3882650436376
// MI455X (gfx1250) — matrix operand order repaired and checked by reading, not compiled, not run on hardware
//
#include <hip/hip_runtime.h>
#include <hip/hip_bf16.h>

// ---------------------------------------------------------------------------
// 2-layer LSTM for MI455X (gfx1250, wave32, WMMA bf16 -> f32, TDM staging)
// B=32, T=256, I=H=1024
// ---------------------------------------------------------------------------

typedef __attribute__((ext_vector_type(16))) __bf16 v16bf;
typedef __attribute__((ext_vector_type(8)))  float  v8f;
typedef __attribute__((ext_vector_type(4)))  unsigned u32x4;
typedef __attribute__((ext_vector_type(8)))  int      i32x8;
typedef __attribute__((ext_vector_type(4)))  int      i32x4;

static constexpr int BB  = 32;
static constexpr int TT  = 256;
static constexpr int HH  = 1024;
static constexpr int M_ROWS = BB * TT;   // 8192
static constexpr int G4H = 4 * HH;       // 4096
static constexpr int NBLK_SCAN = 64;     // persistent blocks in recurrent scan

union Frag16 { v16bf v; uint4 q[2]; };

__device__ inline v8f wmma_bf16(v16bf a, v16bf b, v8f c) {
    return __builtin_amdgcn_wmma_f32_16x16x32_bf16(
        false, a, false, b, (short)0, c, false, false);
}

// A fragment: 16x32 bf16 tile of row-major [?, ldk].
__device__ inline v16bf load_a_frag(const __bf16* base, int m0, int k0, int ldk) {
    const int lane = threadIdx.x & 31;
    const int m  = m0 + (lane & 15);
    const int kh = (lane >> 4) << 3;                // 0 or 8
    const __bf16* p = base + (size_t)m * ldk + k0 + kh;
    Frag16 f;
    f.q[0] = *(const uint4*)(p);
    f.q[1] = *(const uint4*)(p + 16);
    return f.v;
}

// B fragment: 32x16 (KxN) where B[k][n] = W[n][k], W row-major [N, ldk].
__device__ inline v16bf load_b_frag(const __bf16* wbase, int n0, int k0, int ldk) {
    const int lane = threadIdx.x & 31;
    const int n  = n0 + (lane & 15);
    const int kh = (lane >> 4) << 3;                // 0 or 8
    const __bf16* p = wbase + (size_t)n * ldk + k0 + kh;
    Frag16 f;
    f.q[0] = *(const uint4*)(p);
    f.q[1] = *(const uint4*)(p + 16);
    return f.v;
}

// ---------------------------------------------------------------------------
// Tensor Data Mover: 2D tile (nrows x row_elems bf16, row stride in elements)
// from global -> LDS byte offset. D# per CDNA5 ISA.
// Issue from ONE wave only (EXEC ignored by TDM); wait with s_wait_tensorcnt.
// ---------------------------------------------------------------------------
__device__ inline void tdm_load_2d(unsigned lds_off, const void* gptr,
                                   unsigned row_elems, unsigned nrows,
                                   unsigned row_stride_elems) {
    unsigned long long ga = (unsigned long long)gptr;
    u32x4 g0;
    g0.x = 1u;                                        // count=1, user mode
    g0.y = lds_off;                                   // lds_addr (bytes)
    g0.z = (unsigned)(ga & 0xFFFFFFFFu);              // global_addr[95:64]
    g0.w = (unsigned)((ga >> 32) & 0x01FFFFFFu) | (2u << 30); // addr hi | type=2
    i32x8 g1;
    g1[0] = (int)(1u << 16);                          // data_size=1 (2 bytes)
    g1[1] = (int)((row_elems & 0xFFFFu) << 16);       // tensor_dim0[15:0]
    g1[2] = (int)(((row_elems >> 16) & 0xFFFFu) | ((nrows & 0xFFFFu) << 16));
    g1[3] = (int)((row_elems & 0xFFFFu) << 16);       // tile_dim0 (16b)
    g1[4] = (int)(nrows & 0xFFFFu);                   // tile_dim1, tile_dim2=0
    g1[5] = (int)row_stride_elems;                    // tensor_dim0_stride lo
    g1[6] = 0;
    g1[7] = 0;
    i32x4 z4 = {};
#if defined(__clang_major__) && (__clang_major__ >= 23)
    i32x8 z8 = {};
    __builtin_amdgcn_tensor_load_to_lds(g0, g1, z4, z4, z8, 0);
#else
    __builtin_amdgcn_tensor_load_to_lds(g0, g1, z4, z4, 0);
#endif
}

// ---------------------------------------------------------------------------
// f32 -> bf16 conversion
// ---------------------------------------------------------------------------
__global__ __launch_bounds__(256) void cvt_bf16_kernel(const float* __restrict__ in,
                                                       __bf16* __restrict__ out,
                                                       size_t n) {
    size_t i = (size_t)blockIdx.x * blockDim.x + threadIdx.x;
    size_t stride = (size_t)gridDim.x * blockDim.x;
    for (; i < n; i += stride) out[i] = (__bf16)in[i];
}

__global__ void init_barrier_kernel(unsigned* p) {
    p[0] = 0u;  // arrive counter
    p[1] = 0u;  // generation
}

// ---------------------------------------------------------------------------
// D[M,N] = A[M,K](bf16) * W[N,K](bf16)^T + bias[N]   (f32 out)
// Block 256 thr = 8 waves, 64(M) x 128(N) tile. Wave w: M-tiles {2(w&1),+1},
// N-tiles {2(w>>1),+1}: 2x2 WMMA tiles -> 4 wmma per 8 b128 loads.
// ---------------------------------------------------------------------------
__global__ __launch_bounds__(256) void gemm_bias_kernel(
    const __bf16* __restrict__ A, const __bf16* __restrict__ W,
    const float* __restrict__ bias, float* __restrict__ D,
    int M, int N, int K)
{
    const int lane = threadIdx.x & 31;
    const int wave = threadIdx.x >> 5;
    const int m0 = blockIdx.x * 64  + (wave & 1) * 32;
    const int n0 = blockIdx.y * 128 + (wave >> 1) * 32;

    v8f acc00 = {}, acc01 = {}, acc10 = {}, acc11 = {};
    for (int k0 = 0; k0 < K; k0 += 32) {
        v16bf a0 = load_a_frag(A, m0,      k0, K);
        v16bf a1 = load_a_frag(A, m0 + 16, k0, K);
        v16bf b0 = load_b_frag(W, n0,      k0, K);
        v16bf b1 = load_b_frag(W, n0 + 16, k0, K);
        acc00 = wmma_bf16(a0, b0, acc00);
        acc01 = wmma_bf16(a0, b1, acc01);
        acc10 = wmma_bf16(a1, b0, acc10);
        acc11 = wmma_bf16(a1, b1, acc11);
    }
    const int nc0 = n0 + (lane & 15);
    const float bv0 = bias[nc0];
    const float bv1 = bias[nc0 + 16];
    const int mr = m0 + ((lane >> 4) << 3);
    for (int r = 0; r < 8; ++r) {
        D[(size_t)(mr + r)      * N + nc0]      = acc00[r] + bv0;
        D[(size_t)(mr + r)      * N + nc0 + 16] = acc01[r] + bv1;
        D[(size_t)(mr + 16 + r) * N + nc0]      = acc10[r] + bv0;
        D[(size_t)(mr + 16 + r) * N + nc0 + 16] = acc11[r] + bv1;
    }
}

// ---------------------------------------------------------------------------
// Grid-wide barrier (persistent kernel, co-resident blocks)
// ---------------------------------------------------------------------------
__device__ inline void grid_barrier(unsigned* cnt, unsigned* gen, unsigned nblocks) {
    __builtin_amdgcn_fence(__ATOMIC_RELEASE, "agent");
    __syncthreads();
    if (threadIdx.x == 0) {
        unsigned g = __hip_atomic_load(gen, __ATOMIC_RELAXED, __HIP_MEMORY_SCOPE_AGENT);
        unsigned a = __hip_atomic_fetch_add(cnt, 1u, __ATOMIC_ACQ_REL, __HIP_MEMORY_SCOPE_AGENT);
        if (a == nblocks - 1u) {
            __hip_atomic_store(cnt, 0u, __ATOMIC_RELAXED, __HIP_MEMORY_SCOPE_AGENT);
            __hip_atomic_store(gen, g + 1u, __ATOMIC_RELEASE, __HIP_MEMORY_SCOPE_AGENT);
        } else {
            while (__hip_atomic_load(gen, __ATOMIC_ACQUIRE, __HIP_MEMORY_SCOPE_AGENT) == g) {
                __builtin_amdgcn_s_sleep(2);
            }
        }
    }
    __syncthreads();
    __builtin_amdgcn_fence(__ATOMIC_ACQUIRE, "agent");
}

// ---------------------------------------------------------------------------
// Recurrent scan. 64 persistent blocks x 256 threads.
// Block b owns h/c columns [b*16, b*16+16).
// LDS: Whh slice 64x1024 bf16 (128KB, TDM once) + h 32x1024 bf16 (64KB, TDM
// per step) + pre 32x64 f32 (8KB) + cell 32x16 f32 (2KB) = ~202KB / 320KB.
// ---------------------------------------------------------------------------
static constexpr unsigned SOFF_W   = 0;                         // 131072 B
static constexpr unsigned SOFF_H   = 64 * 1024 * 2;             // +65536 B
static constexpr unsigned SOFF_PRE = SOFF_H + 32 * 1024 * 2;    // +8192 B
static constexpr unsigned SOFF_C   = SOFF_PRE + 32 * 64 * 4;    // +2048 B
static constexpr unsigned SMEM_SCAN = SOFF_C + 32 * 16 * 4;

__global__ __launch_bounds__(256) void lstm_scan_kernel(
    const float* __restrict__ ih,      // [B*T, 4H]  row = b*T + t
    const __bf16* __restrict__ Whh,    // [4H, H] row-major bf16
    __bf16* __restrict__ hping,        // [B, H] bf16
    __bf16* __restrict__ hpong,        // [B, H] bf16
    __bf16* __restrict__ ybf,          // [B, T, H] bf16 (next layer input)
    float*  __restrict__ yf32,         // [B, T, H] f32 final output (or null)
    unsigned* __restrict__ bar)
{
    extern __shared__ char smem[];
    __bf16* sW   = (__bf16*)(smem + SOFF_W);    // [64][1024]  lr = gate*16 + j
    __bf16* sH   = (__bf16*)(smem + SOFF_H);    // [32][1024]
    float*  sPre = (float*)(smem + SOFF_PRE);   // [32][64]
    float*  sC   = (float*)(smem + SOFF_C);     // [32][16]

    const int tid  = threadIdx.x;
    const int lane = tid & 31;
    const int wave = tid >> 5;           // 0..7
    const int blk  = blockIdx.x;         // 0..63
    const int nt   = wave & 3;           // N tile == gate
    const int mt   = wave >> 2;          // M tile (0/1)

    // Whh slice: 4 chunks of 16 contiguous rows (gate*H + blk*16 ..+16),
    // chunk stride = H rows -> one 2D TDM: 4 rows of 16384 elems, stride H*H.
    if (wave == 0) {
        tdm_load_2d(SOFF_W, Whh + (size_t)blk * 16 * HH,
                    /*row_elems=*/16 * HH, /*nrows=*/4,
                    /*row_stride_elems=*/(unsigned)HH * HH);
    }
    // init cell state + zero our h columns in ping/pong
    for (int i = tid; i < BB * 16; i += 256) {
        int m = i >> 4, j = i & 15;
        sC[i] = 0.0f;
        hping[(size_t)m * HH + blk * 16 + j] = (__bf16)0.0f;
        hpong[(size_t)m * HH + blk * 16 + j] = (__bf16)0.0f;
    }
    if (wave == 0) __builtin_amdgcn_s_wait_tensorcnt(0);
    grid_barrier(bar, bar + 1, NBLK_SCAN);

    const int lcol  = nt * 16 + (lane & 15);                  // col in sPre
    const int ncg   = nt * HH + blk * 16 + (lane & 15);       // col in ih
    const int mbase = mt * 16 + ((lane >> 4) << 3);

    for (int t = 0; t < TT; ++t) {
        const __bf16* hin  = (t & 1) ? hpong : hping;
        __bf16*       hout = (t & 1) ? hping : hpong;

        // stage h_t (64KB) into LDS via TDM
        if (wave == 0) {
            tdm_load_2d(SOFF_H, hin, BB * HH, 1, BB * HH);
            __builtin_amdgcn_s_wait_tensorcnt(0);
        }
        __syncthreads();

        // pre tile: 16x16 per wave, K = 1024, all operands in LDS
        v8f acc = {};
        for (int k0 = 0; k0 < HH; k0 += 32) {
            v16bf a = load_a_frag(sH, mt * 16, k0, HH);
            v16bf b = load_b_frag(sW, nt * 16, k0, HH);
            acc = wmma_bf16(a, b, acc);
        }
        for (int r = 0; r < 8; ++r) {
            int m = mbase + r;
            sPre[m * 64 + lcol] = acc[r] + ih[((size_t)m * TT + t) * G4H + ncg];
        }
        __syncthreads();

        // gates + state update: 32 rows x 16 cols, 2 per thread
        for (int i = tid; i < BB * 16; i += 256) {
            int m = i >> 4, j = i & 15;
            float iv = 1.0f / (1.0f + __expf(-sPre[m * 64 + 0 * 16 + j]));
            float fv = 1.0f / (1.0f + __expf(-sPre[m * 64 + 1 * 16 + j]));
            float gv = tanhf(sPre[m * 64 + 2 * 16 + j]);
            float ov = 1.0f / (1.0f + __expf(-sPre[m * 64 + 3 * 16 + j]));
            float cn = fv * sC[i] + iv * gv;
            float hn = ov * tanhf(cn);
            sC[i] = cn;
            hout[(size_t)m * HH + blk * 16 + j] = (__bf16)hn;
            ybf[((size_t)m * TT + t) * HH + blk * 16 + j] = (__bf16)hn;
            if (yf32)
                yf32[((size_t)m * TT + t) * HH + blk * 16 + j] = hn;
        }
        grid_barrier(bar, bar + 1, NBLK_SCAN);   // publish h_t to all blocks
    }
}

// ---------------------------------------------------------------------------
// Workspace layout (bytes)
// ---------------------------------------------------------------------------
static constexpr size_t OFF_BAR   = 0;
static constexpr size_t OFF_HPING = 256;
static constexpr size_t OFF_HPONG = OFF_HPING + (size_t)BB * HH * 2;
static constexpr size_t OFF_XB    = OFF_HPONG + (size_t)BB * HH * 2;
static constexpr size_t SZ_X      = (size_t)M_ROWS * HH * 2;           // 16 MB
static constexpr size_t SZ_W      = (size_t)G4H * HH * 2;              // 8 MB
static constexpr size_t OFF_W0I   = OFF_XB  + SZ_X;
static constexpr size_t OFF_W0H   = OFF_W0I + SZ_W;
static constexpr size_t OFF_W1I   = OFF_W0H + SZ_W;
static constexpr size_t OFF_W1H   = OFF_W1I + SZ_W;
static constexpr size_t OFF_Y0B   = OFF_W1H + SZ_W;                    // bf16 y0
static constexpr size_t OFF_IH    = OFF_Y0B + SZ_X;                    // f32

extern "C" void kernel_launch(void* const* d_in, const int* in_sizes, int n_in,
                              void* d_out, int out_size, void* d_ws, size_t ws_size,
                              hipStream_t stream) {
    (void)in_sizes; (void)n_in; (void)out_size; (void)ws_size;
    const float* x    = (const float*)d_in[0];
    const float* Wih0 = (const float*)d_in[1];
    const float* Whh0 = (const float*)d_in[2];
    const float* b0   = (const float*)d_in[3];
    const float* Wih1 = (const float*)d_in[4];
    const float* Whh1 = (const float*)d_in[5];
    const float* b1   = (const float*)d_in[6];
    float* out = (float*)d_out;

    char* ws = (char*)d_ws;
    unsigned* bar   = (unsigned*)(ws + OFF_BAR);
    __bf16* hping   = (__bf16*)(ws + OFF_HPING);
    __bf16* hpong   = (__bf16*)(ws + OFF_HPONG);
    __bf16* xb      = (__bf16*)(ws + OFF_XB);
    __bf16* w0i     = (__bf16*)(ws + OFF_W0I);
    __bf16* w0h     = (__bf16*)(ws + OFF_W0H);
    __bf16* w1i     = (__bf16*)(ws + OFF_W1I);
    __bf16* w1h     = (__bf16*)(ws + OFF_W1H);
    __bf16* y0b     = (__bf16*)(ws + OFF_Y0B);
    float*  ihbuf   = (float*)(ws + OFF_IH);

    init_barrier_kernel<<<1, 1, 0, stream>>>(bar);

    cvt_bf16_kernel<<<2048, 256, 0, stream>>>(x,    xb,  (size_t)M_ROWS * HH);
    cvt_bf16_kernel<<<2048, 256, 0, stream>>>(Wih0, w0i, (size_t)G4H * HH);
    cvt_bf16_kernel<<<2048, 256, 0, stream>>>(Whh0, w0h, (size_t)G4H * HH);
    cvt_bf16_kernel<<<2048, 256, 0, stream>>>(Wih1, w1i, (size_t)G4H * HH);
    cvt_bf16_kernel<<<2048, 256, 0, stream>>>(Whh1, w1h, (size_t)G4H * HH);

    dim3 ggrid(M_ROWS / 64, G4H / 128);

    // Layer 0
    gemm_bias_kernel<<<ggrid, 256, 0, stream>>>(xb, w0i, b0, ihbuf, M_ROWS, G4H, HH);
    lstm_scan_kernel<<<NBLK_SCAN, 256, SMEM_SCAN, stream>>>(ihbuf, w0h, hping, hpong,
                                                            y0b, nullptr, bar);
    // Layer 1
    gemm_bias_kernel<<<ggrid, 256, 0, stream>>>(y0b, w1i, b1, ihbuf, M_ROWS, G4H, HH);
    lstm_scan_kernel<<<NBLK_SCAN, 256, SMEM_SCAN, stream>>>(ihbuf, w1h, hping, hpong,
                                                            y0b /*scratch*/, out, bar);
}
